// FadingBlock_48722109005855
// MI455X (gfx1250) — matrix operand order repaired and checked by reading, not compiled, not run on hardware
//
#include <hip/hip_runtime.h>
#include <hip/hip_bf16.h>
#include <math.h>

// ---------------------------------------------------------------------------
// Problem constants (reference: B=2, T=2048, C=1024, H=16, HD=64)
#define BB 2
#define TT 2048
#define CC 1024
#define HH 16
#define HD 64
#define FF (4*CC)
#define MM (BB*TT)

typedef __attribute__((ext_vector_type(16))) _Float16 v16h;
typedef __attribute__((ext_vector_type(8)))  float    v8f;

// ---------------------------------------------------------------------------
// WMMA helpers (wave32, v_wmma_f32_16x16x32_f16)

__device__ inline v8f wmma32(v16h a, v16h b, v8f c) {
#if defined(__gfx1250__)
  return __builtin_amdgcn_wmma_f32_16x16x32_f16(false, a, false, b, (short)0, c,
                                                false, false);
#else
  return c;
#endif
}

// A fragment: 16x32 f16, row-major source with leading dim lda.
// lane<16 : row=lane,    halves j<8 -> k=j,     j>=8 -> k=16+(j-8)
// lane>=16: row=lane-16, halves j<8 -> k=8+j,   j>=8 -> k=24+(j-8)
__device__ inline v16h load_a_frag(const _Float16* A, int lda) {
  int lane = threadIdx.x & 31;
  const _Float16* p = A + (size_t)(lane & 15) * lda + ((lane >> 4) * 8);
  v16h f;
#pragma unroll
  for (int j = 0; j < 8; ++j) { f[j] = p[j]; f[8 + j] = p[16 + j]; }
  return f;
}

// B fragment: 32x16 f16 from Bt = B^T stored row-major [N x K], leading dim ldb.
// lane gives column n = lane&15; j<8 -> k = 8*(lane>>4) + j, j>=8 -> k = 16 + 8*(lane>>4) + (j-8)
__device__ inline v16h load_b_frag(const _Float16* Bt, int ldb) {
  int lane = threadIdx.x & 31;
  const _Float16* p = Bt + (size_t)(lane & 15) * ldb + ((lane >> 4) * 8);
  v16h f;
#pragma unroll
  for (int j = 0; j < 8; ++j) { f[j] = p[j]; f[8 + j] = p[16 + j]; }
  return f;
}

// C/D layout: col n = lane&15, rows m = r + 8*(lane>>4), r = 0..7

// ---------------------------------------------------------------------------
// CDNA5 async global->LDS copy (ASYNCcnt-tracked), GV addressing.
// vdst = 32-bit LDS byte offset, vaddr = 64-bit global address pair.
__device__ inline void async_copy_b128(unsigned int lds_off, const void* g) {
#if defined(__gfx1250__)
  asm volatile("global_load_async_to_lds_b128 %0, %1, off"
               :: "v"(lds_off), "v"((unsigned long long)(uintptr_t)g)
               : "memory");
#else
  (void)lds_off; (void)g;
#endif
}

__device__ inline void wait_async_le2() {
#if defined(__gfx1250__)
  asm volatile("s_wait_asynccnt 0x2" ::: "memory");
#endif
}
__device__ inline void wait_async_0() {
#if defined(__gfx1250__)
  asm volatile("s_wait_asynccnt 0x0" ::: "memory");
#endif
}

// Stage a 64-row x 32-half tile of Bt (row stride ldk halves) into LDS
// (row stride 32 halves). 128 threads, 2 x 16B chunks each (256 chunks).
__device__ inline void stage_b_tile(const _Float16* src, int ldk,
                                    _Float16* dst) {
  int tid = threadIdx.x;
#pragma unroll
  for (int i = 0; i < 2; ++i) {
    int c = tid * 2 + i;           // 0..255
    int row = c >> 2;              // 64 rows
    int q = (c & 3) * 8;           // 4 chunks of 8 halves per row
    const _Float16* g = src + (size_t)row * ldk + q;
    unsigned int l = (unsigned int)(uintptr_t)(dst + row * 32 + q);
    async_copy_b128(l, g);
  }
}

// ---------------------------------------------------------------------------
// LayerNorm: one block per row, fp32 stats, f16 output
__global__ __launch_bounds__(256) void ln_kernel(const float* __restrict__ x,
                                                 const float* __restrict__ w,
                                                 _Float16* __restrict__ out) {
  __shared__ float s1[256], s2[256];
  int row = blockIdx.x;
  const float* xr = x + (size_t)row * CC;
  float a = 0.f, b = 0.f;
  for (int i = threadIdx.x; i < CC; i += 256) { float v = xr[i]; a += v; b += v * v; }
  s1[threadIdx.x] = a; s2[threadIdx.x] = b;
  __syncthreads();
  for (int off = 128; off > 0; off >>= 1) {
    if ((int)threadIdx.x < off) {
      s1[threadIdx.x] += s1[threadIdx.x + off];
      s2[threadIdx.x] += s2[threadIdx.x + off];
    }
    __syncthreads();
  }
  float mu = s1[0] * (1.f / CC);
  float var = s2[0] * (1.f / CC) - mu * mu;
  float inv = rsqrtf(var + 1e-5f);
  for (int i = threadIdx.x; i < CC; i += 256)
    out[(size_t)row * CC + i] = (_Float16)((xr[i] - mu) * inv * w[i]);
}

// ---------------------------------------------------------------------------
// Weight conversion kernels
__global__ void conv_f16_kernel(const float* __restrict__ in,
                                _Float16* __restrict__ out, size_t n) {
  for (size_t i = (size_t)blockIdx.x * blockDim.x + threadIdx.x; i < n;
       i += (size_t)gridDim.x * blockDim.x)
    out[i] = (_Float16)in[i];
}

// (H, C, HD) f32  ->  (H, HD, C) f16
__global__ void conv_trans_hcd_kernel(const float* __restrict__ in,
                                      _Float16* __restrict__ out) {
  size_t i = (size_t)blockIdx.x * blockDim.x + threadIdx.x;
  size_t total = (size_t)HH * HD * CC;
  if (i >= total) return;
  int c = i % CC;
  int d = (i / CC) % HD;
  int h = i / ((size_t)CC * HD);
  out[i] = (_Float16)in[((size_t)h * CC + c) * HD + d];
}

// ---------------------------------------------------------------------------
// QKV: grid = B*H*(T/16), block = 128 (4 waves; wave w -> d-tile w)
// q scaled by C^-0.5, v stored transposed (B,H,HD,T)
__global__ __launch_bounds__(128) void qkv_kernel(
    const _Float16* __restrict__ h1,                       // (B,T,C) f16
    const _Float16* __restrict__ WqT, const _Float16* __restrict__ WkT,
    const _Float16* __restrict__ WvT,                      // (H,HD,C) f16
    const float* __restrict__ bq, const float* __restrict__ bk,
    const float* __restrict__ bv,                          // (H,HD)
    _Float16* __restrict__ qo, _Float16* __restrict__ ko,  // (B,H,T,HD)
    _Float16* __restrict__ vTo)                            // (B,H,HD,T)
{
  int wave = threadIdx.x >> 5;
  int lane = threadIdx.x & 31;
  int tpb = TT / 16;
  int tt = blockIdx.x % tpb;
  int bh = blockIdx.x / tpb;
  int b = bh >> 4, h = bh & 15;

  const _Float16* A  = h1 + ((size_t)b * TT + tt * 16) * CC;
  const _Float16* Bq = WqT + ((size_t)h * HD + wave * 16) * CC;
  const _Float16* Bk = WkT + ((size_t)h * HD + wave * 16) * CC;
  const _Float16* Bv = WvT + ((size_t)h * HD + wave * 16) * CC;

  v8f aq = {}, ak = {}, av = {};
  v16h a = load_a_frag(A, CC);
  for (int kk = 0; kk < CC; kk += 32) {
    v16h a_next = a;
    if (kk + 32 < CC) a_next = load_a_frag(A + kk + 32, CC);
    aq = wmma32(a, load_b_frag(Bq + kk, CC), aq);
    ak = wmma32(a, load_b_frag(Bk + kk, CC), ak);
    av = wmma32(a, load_b_frag(Bv + kk, CC), av);
    a = a_next;
  }
  int n = lane & 15, mb = (lane >> 4) * 8;
  int d = wave * 16 + n;
  size_t base = ((size_t)bh * TT + tt * 16) * HD;
  float bqv = bq[h * HD + d], bkv = bk[h * HD + d], bvv = bv[h * HD + d];
#pragma unroll
  for (int r = 0; r < 8; ++r) {
    int t = mb + r;
    qo[base + (size_t)t * HD + d] = (_Float16)((aq[r] + bqv) * 0.03125f);
    ko[base + (size_t)t * HD + d] = (_Float16)(ak[r] + bkv);
    vTo[(size_t)bh * HD * TT + (size_t)d * TT + (tt * 16 + t)] =
        (_Float16)(av[r] + bvv);
  }
}

// ---------------------------------------------------------------------------
// Flash-style causal attention: grid = B*H*(T/16), block = 32 (one wave)
__global__ __launch_bounds__(32) void attn_kernel(
    const _Float16* __restrict__ q,   // (B,H,T,HD), pre-scaled
    const _Float16* __restrict__ k,   // (B,H,T,HD)
    const _Float16* __restrict__ vT,  // (B,H,HD,T)
    _Float16* __restrict__ o)         // (B,T,C) f16, c = h*HD + d
{
  __shared__ _Float16 ptile[16 * 32];
  int qt = blockIdx.x % (TT / 16);
  int bh = blockIdx.x / (TT / 16);
  int b = bh >> 4, h = bh & 15;
  int lane = threadIdx.x;
  int n = lane & 15, mb = (lane >> 4) * 8;

  const _Float16* qb = q + ((size_t)bh * TT + qt * 16) * HD;
  const _Float16* kb = k + (size_t)bh * TT * HD;
  const _Float16* vb = vT + (size_t)bh * HD * TT;

  v16h qa0 = load_a_frag(qb, HD);
  v16h qa1 = load_a_frag(qb + 32, HD);

  float mrun[8], lrun[8];
  v8f oacc[4] = {};
#pragma unroll
  for (int r = 0; r < 8; ++r) { mrun[r] = -1e30f; lrun[r] = 0.f; }

  int npair = qt / 2 + 1;  // s-tile pairs covering s <= qt*16+15
  for (int st = 0; st < npair; ++st) {
    int s0 = st * 32;
    v8f att[2];
#pragma unroll
    for (int j = 0; j < 2; ++j) {
      v8f acc = {};
      acc = wmma32(qa0, load_b_frag(kb + (size_t)(s0 + 16 * j) * HD, HD), acc);
      acc = wmma32(qa1, load_b_frag(kb + (size_t)(s0 + 16 * j) * HD + 32, HD), acc);
      att[j] = acc;
    }
    // online softmax over the 32 columns of this pair
#pragma unroll
    for (int r = 0; r < 8; ++r) {
      int tg = qt * 16 + mb + r;
      float a0 = (s0 + n)      <= tg ? att[0][r] : -1e30f;
      float a1 = (s0 + 16 + n) <= tg ? att[1][r] : -1e30f;
      float mx = fmaxf(a0, a1);
      for (int off = 1; off < 16; off <<= 1) mx = fmaxf(mx, __shfl_xor(mx, off, 32));
      float mnew = fmaxf(mrun[r], mx);
      float scale = __expf(mrun[r] - mnew);
      mrun[r] = mnew;
      lrun[r] *= scale;
#pragma unroll
      for (int nt = 0; nt < 4; ++nt) oacc[nt][r] *= scale;
      float p0 = __expf(a0 - mnew), p1 = __expf(a1 - mnew);
      float ps = p0 + p1;
      for (int off = 1; off < 16; off <<= 1) ps += __shfl_xor(ps, off, 32);
      lrun[r] += ps;
      ptile[(mb + r) * 32 + n]      = (_Float16)p0;
      ptile[(mb + r) * 32 + 16 + n] = (_Float16)p1;
    }
    __syncthreads();
    v16h pa = load_a_frag(ptile, 32);  // reshape C-layout -> A-fragment via LDS
#pragma unroll
    for (int nt = 0; nt < 4; ++nt)
      oacc[nt] = wmma32(pa, load_b_frag(vb + (size_t)(16 * nt) * TT + s0, TT),
                        oacc[nt]);
    __syncthreads();
  }
  _Float16* ob = o + ((size_t)b * TT + qt * 16) * CC + h * HD;
#pragma unroll
  for (int nt = 0; nt < 4; ++nt)
#pragma unroll
    for (int r = 0; r < 8; ++r)
      ob[(size_t)(mb + r) * CC + nt * 16 + n] =
          (_Float16)(oacc[nt][r] / lrun[r]);
}

// ---------------------------------------------------------------------------
// Staged GEMM with async-LDS double buffering of the B tile.
// Block = 128 threads (4 waves). Block tile: 64(m) x 64(n); wave w owns rows
// [m0+16w, m0+16w+16). B tile (64n x 32k) staged via ASYNCcnt-tracked
// global_load_async_to_lds_b128, double buffered; fragments re-read as
// ds_load_b128 and shared by all 4 waves. A fragments pipelined in registers.
// MODE 0: out = resid + A@B (f32)                       [Wo projection]
// MODE 1: outh = gelu(A@B) (f16)                        [FFN1]
// MODE 2: out = resid + A@B (f32), x2t = (f16) out^T    [FFN2]
template <int MODE>
__global__ __launch_bounds__(128) void gemm_staged_kernel(
    const _Float16* __restrict__ A, const _Float16* __restrict__ Bt,
    const float* __restrict__ resid, float* __restrict__ outf,
    _Float16* __restrict__ outh, _Float16* __restrict__ x2t, int N, int K) {
  __shared__ _Float16 bbuf[2][64 * 32];
  int wave = threadIdx.x >> 5;
  int lane = threadIdx.x & 31;
  int m0 = blockIdx.x * 64 + wave * 16;
  int n0 = blockIdx.y * 64;

  const _Float16* Arow = A + (size_t)m0 * K;
  const _Float16* Bt0 = Bt + (size_t)n0 * K;

  v8f acc[4] = {};
  // preload stage 0 (async) and first A fragment (registers)
  stage_b_tile(Bt0, K, bbuf[0]);
  v16h a = load_a_frag(Arow, K);

  int cur = 0;
  for (int kk = 0; kk < K; kk += 32) {
    v16h a_next = a;
    if (kk + 32 < K) {
      stage_b_tile(Bt0 + kk + 32, K, bbuf[cur ^ 1]);  // prefetch next B tile
      a_next = load_a_frag(Arow + kk + 32, K);        // prefetch next A frag
      __builtin_prefetch(Arow + kk + 64, 0, 1);       // global_prefetch_b8
      wait_async_le2();   // own 2 prefetch ops may remain in flight
    } else {
      wait_async_0();
    }
    __syncthreads();      // B tile `cur` complete for all waves
#pragma unroll
    for (int nt = 0; nt < 4; ++nt)
      acc[nt] = wmma32(a, load_b_frag(bbuf[cur] + 16 * nt * 32, 32), acc[nt]);
    __syncthreads();      // all waves done reading before next overwrite
    a = a_next;
    cur ^= 1;
  }

  int n = lane & 15, mb = (lane >> 4) * 8;
#pragma unroll
  for (int nt = 0; nt < 4; ++nt)
#pragma unroll
    for (int r = 0; r < 8; ++r) {
      int row = m0 + mb + r;
      int col = n0 + 16 * nt + n;
      size_t idx = (size_t)row * N + col;
      if constexpr (MODE == 1) {
        float v = acc[nt][r];
        outh[idx] = (_Float16)(0.5f * v * (1.0f + erff(v * 0.70710678118f)));
      } else {
        float v = resid[idx] + acc[nt][r];
        outf[idx] = v;
        if constexpr (MODE == 2) {
          int b = row >> 11, t = row & (TT - 1);
          x2t[((size_t)b * CC + col) * TT + t] = (_Float16)v;
        }
      }
    }
}

// ---------------------------------------------------------------------------
// y = fW @ x2[:, t_off:t_off+K, :]  -> d_out rows [o_off, o_off+O)
// grid: (O/16, C/64, B), block = 32
__global__ __launch_bounds__(32) void fading_gemm_kernel(
    const _Float16* __restrict__ fW,   // (O,K) f16 row-major
    const _Float16* __restrict__ x2t,  // (B,C,T) f16
    float* __restrict__ out,           // (B,1024,C) f32
    int K, int t_off, int o_off) {
  int b = blockIdx.z;
  int m0 = blockIdx.x * 16, n0 = blockIdx.y * 64;
  const _Float16* Bt = x2t + (size_t)b * CC * TT + t_off;
  v8f acc[4] = {};
  v16h a = load_a_frag(fW + (size_t)m0 * K, K);
  for (int kk = 0; kk < K; kk += 32) {
    v16h a_next = a;
    if (kk + 32 < K) a_next = load_a_frag(fW + (size_t)m0 * K + kk + 32, K);
#pragma unroll
    for (int nt = 0; nt < 4; ++nt)
      acc[nt] = wmma32(a, load_b_frag(Bt + (size_t)(n0 + 16 * nt) * TT + kk, TT),
                       acc[nt]);
    a = a_next;
  }
  int lane = threadIdx.x, n = lane & 15, mb = (lane >> 4) * 8;
#pragma unroll
  for (int nt = 0; nt < 4; ++nt)
#pragma unroll
    for (int r = 0; r < 8; ++r)
      out[((size_t)b * 1024 + o_off + m0 + mb + r) * CC + n0 + 16 * nt + n] =
          acc[nt][r];
}

// ---------------------------------------------------------------------------
extern "C" void kernel_launch(void* const* d_in, const int* in_sizes, int n_in,
                              void* d_out, int out_size, void* d_ws,
                              size_t ws_size, hipStream_t stream) {
  const float* x    = (const float*)d_in[0];
  const float* ln1w = (const float*)d_in[1];
  const float* Wq   = (const float*)d_in[2];
  const float* bq   = (const float*)d_in[3];
  const float* Wk   = (const float*)d_in[4];
  const float* bk   = (const float*)d_in[5];
  const float* Wv   = (const float*)d_in[6];
  const float* bv   = (const float*)d_in[7];
  const float* Wo   = (const float*)d_in[8];
  const float* ln2w = (const float*)d_in[9];
  const float* W1   = (const float*)d_in[10];
  const float* W2   = (const float*)d_in[11];
  const float* fW1  = (const float*)d_in[12];
  const float* fW2  = (const float*)d_in[13];
  const float* fW3  = (const float*)d_in[14];
  float* out = (float*)d_out;

  // workspace carve-up
  char* w = (char*)d_ws;
  size_t off = 0;
  auto alloc = [&](size_t bytes) {
    void* p = w + off;
    off = (off + bytes + 255) & ~(size_t)255;
    return p;
  };
  _Float16* h1   = (_Float16*)alloc((size_t)MM * CC * 2);
  _Float16* WqT  = (_Float16*)alloc((size_t)HH * HD * CC * 2);
  _Float16* WkT  = (_Float16*)alloc((size_t)HH * HD * CC * 2);
  _Float16* WvT  = (_Float16*)alloc((size_t)HH * HD * CC * 2);
  _Float16* Woh  = (_Float16*)alloc((size_t)CC * CC * 2);
  _Float16* W1h  = (_Float16*)alloc((size_t)FF * CC * 2);
  _Float16* W2h  = (_Float16*)alloc((size_t)CC * FF * 2);
  _Float16* fW1h = (_Float16*)alloc((size_t)128 * 768 * 2);
  _Float16* fW2h = (_Float16*)alloc((size_t)128 * 384 * 2);
  _Float16* fW3h = (_Float16*)alloc((size_t)256 * 384 * 2);
  _Float16* qh   = (_Float16*)alloc((size_t)BB * HH * TT * HD * 2);
  _Float16* kh   = (_Float16*)alloc((size_t)BB * HH * TT * HD * 2);
  _Float16* vTh  = (_Float16*)alloc((size_t)BB * HH * HD * TT * 2);
  _Float16* o16  = (_Float16*)alloc((size_t)MM * CC * 2);
  float*    x1   = (float*)alloc((size_t)MM * CC * 4);
  _Float16* h2   = (_Float16*)alloc((size_t)MM * CC * 2);
  _Float16* g16  = (_Float16*)alloc((size_t)MM * FF * 2);
  float*    x2   = (float*)alloc((size_t)MM * CC * 4);
  _Float16* x2t  = (_Float16*)alloc((size_t)BB * CC * TT * 2);
  (void)ws_size; (void)n_in; (void)in_sizes; (void)out_size;

  // 1) weight conversions
  {
    size_t tw = (size_t)HH * HD * CC;
    int blks = (int)((tw + 255) / 256);
    conv_trans_hcd_kernel<<<blks, 256, 0, stream>>>(Wq, WqT);
    conv_trans_hcd_kernel<<<blks, 256, 0, stream>>>(Wk, WkT);
    conv_trans_hcd_kernel<<<blks, 256, 0, stream>>>(Wv, WvT);
    conv_f16_kernel<<<2048, 256, 0, stream>>>(Wo, Woh, (size_t)CC * CC);
    conv_f16_kernel<<<4096, 256, 0, stream>>>(W1, W1h, (size_t)FF * CC);
    conv_f16_kernel<<<4096, 256, 0, stream>>>(W2, W2h, (size_t)CC * FF);
    conv_f16_kernel<<<384, 256, 0, stream>>>(fW1, fW1h, (size_t)128 * 768);
    conv_f16_kernel<<<192, 256, 0, stream>>>(fW2, fW2h, (size_t)128 * 384);
    conv_f16_kernel<<<384, 256, 0, stream>>>(fW3, fW3h, (size_t)256 * 384);
  }

  // 2) LN1
  ln_kernel<<<MM, 256, 0, stream>>>(x, ln1w, h1);

  // 3) QKV projections
  qkv_kernel<<<BB * HH * (TT / 16), 128, 0, stream>>>(h1, WqT, WkT, WvT, bq, bk,
                                                      bv, qh, kh, vTh);

  // 4) causal flash attention
  attn_kernel<<<BB * HH * (TT / 16), 32, 0, stream>>>(qh, kh, vTh, o16);

  // 5) Wo projection + residual -> x1   (async-LDS staged GEMM)
  gemm_staged_kernel<0><<<dim3(MM / 64, CC / 64), 128, 0, stream>>>(
      o16, Woh, x, x1, nullptr, nullptr, CC, CC);

  // 6) LN2
  ln_kernel<<<MM, 256, 0, stream>>>(x1, ln2w, h2);

  // 7) FFN1 + GELU
  gemm_staged_kernel<1><<<dim3(MM / 64, FF / 64), 128, 0, stream>>>(
      h2, W1h, nullptr, nullptr, g16, nullptr, FF, CC);

  // 8) FFN2 + residual -> x2 (f32) and x2t (f16 transposed)
  gemm_staged_kernel<2><<<dim3(MM / 64, CC / 64), 128, 0, stream>>>(
      g16, W2h, x1, x2, nullptr, x2t, CC, FF);

  // 9) fading GEMMs straight into d_out
  fading_gemm_kernel<<<dim3(128 / 16, CC / 64, BB), 32, 0, stream>>>(
      fW1h, x2t, out, 768, 0, 0);
  fading_gemm_kernel<<<dim3(128 / 16, CC / 64, BB), 32, 0, stream>>>(
      fW2h, x2t, out, 384, 768, 128);
  fading_gemm_kernel<<<dim3(256 / 16, CC / 64, BB), 32, 0, stream>>>(
      fW3h, x2t, out, 384, 1152, 256);

  // 10) x4 passthrough: x2[:, 1536:2048, :] -> out rows [512,1024)
  for (int b = 0; b < BB; ++b) {
    hipMemcpyAsync(out + ((size_t)b * 1024 + 512) * CC,
                   x2 + ((size_t)b * TT + 1536) * CC,
                   (size_t)512 * CC * sizeof(float), hipMemcpyDeviceToDevice,
                   stream);
  }
}
